// PretrainModel_57827439674063
// MI455X (gfx1250) — compile-verified, matrix operand order checked by reading, not run on hardware
//
#include <hip/hip_runtime.h>
#include <hip/hip_bf16.h>

// ---------------------------------------------------------------------------
// MI455X (gfx1250) implementation. All dense algebra is lowered to
// v_wmma_f32_16x16x32_f16 (f16 in, f32 accum). LDS tiles are stored in
// fragment-swizzled order so each 16x32/32x16 fragment is two ds_load_b128
// per lane; staging is global v8h load + ds_store_b128. 8 waves per block,
// each wave computes a 32x32 register tile (2x2 WMMA accumulators).
// ---------------------------------------------------------------------------

typedef _Float16 v16h __attribute__((ext_vector_type(16)));
typedef _Float16 v8h  __attribute__((ext_vector_type(8)));
typedef float    v8f  __attribute__((ext_vector_type(8)));

#define GF_BIAS  1
#define GF_RELU  2
#define GF_RESID 4

#define BB      64
#define SEQL    512
#define FEATD   32
#define HDIM    512
#define NLAYER  4
#define NHEAD   8
#define FFDIM   2048
#define NPATCH  128
#define NTOK    (BB * NPATCH)      // 8192
#define THDIM   1024
#define DH      (HDIM / NHEAD)     // 64
#define NN2     (NPATCH * NPATCH)  // 16384

__device__ __forceinline__ v16h frag16(const _Float16* p) {
    v8h lo = *(const v8h*)p;
    v8h hi = *(const v8h*)(p + 8);
    return __builtin_shufflevector(lo, hi, 0, 1, 2, 3, 4, 5, 6, 7,
                                   8, 9, 10, 11, 12, 13, 14, 15);
}

__device__ __forceinline__ v8f wmma_step(v16h a, v16h b, v8f c) {
    return __builtin_amdgcn_wmma_f32_16x16x32_f16(false, a, false, b,
                                                  (short)0, c, false, false);
}

// ============================== WMMA GEMM ==================================
// C[M,N](f32) (+bias)(+resid)(relu) = A[M,K](f16) x B[K,N or N,K](f16)
// batched via blockIdx.z with element strides sA/sB/sC. Ch optional f16 copy.
// Requires K % 32 == 0 (true for every GEMM in this model).
__global__ __launch_bounds__(256) void wmma_gemm(
    const _Float16* __restrict__ A, const _Float16* __restrict__ B,
    float* __restrict__ C, _Float16* __restrict__ Ch,
    const float* __restrict__ bias, const float* __restrict__ resid,
    int M, int N, int K, int lda, int ldb, int ldc,
    long sA, long sB, long sC, int transB, int flags)
{
    const int BM = 128, BN = 64, BK = 32;
    // fragment-swizzled tiles: [strip][lane][16 halves (+8 pad)]
    __shared__ _Float16 As[8][32][24];   // 12 KB
    __shared__ _Float16 Bs[4][32][24];   //  6 KB

    const int tid  = threadIdx.x;
    const int lane = tid & 31;
    const int wv   = tid >> 5;                 // 8 waves
    const int wr   = (wv & 3) * 2;             // 2 row strips per wave
    const int wc   = (wv >> 2) * 2;            // 2 col strips per wave
    const int blockRow = blockIdx.y * BM;
    const int blockCol = blockIdx.x * BN;
    const long zA = (long)blockIdx.z * sA;
    const long zB = (long)blockIdx.z * sB;
    const long zC = (long)blockIdx.z * sC;

    v8f acc[2][2] = {};

    // A staging map: thread -> row ar, two 8-half k-groups
    const int ar    = tid >> 1;                // 0..127
    const int astrip = ar >> 4, alrow = ar & 15;
    // B staging map: thread -> col nB, one 8-half k-group
    const int nB = tid >> 2;                   // 0..63
    const int kg = tid & 3;                    // 0..3 (kb = kg*8)
    const int ctB   = nB >> 4;
    const int laneB = (nB & 15) + 16 * (kg & 1);
    const int j0B   = (kg >> 1) * 8;

    for (int k0 = 0; k0 < K; k0 += BK) {
        // ---- stage A tile (128 x 32), swizzled, vectorized ----
        {
            const int gm = blockRow + ar;
            #pragma unroll
            for (int g = 0; g < 2; ++g) {
                const int kb = (tid & 1) * 16 + g * 8;
                const int lslot = alrow + 16 * ((kb >> 3) & 1);
                const int j0 = (kb >> 4) << 3;
                v8h v;
                if (gm < M) {
                    v = *(const v8h*)&A[zA + (long)gm * lda + k0 + kb];
                } else {
                    #pragma unroll
                    for (int j = 0; j < 8; ++j) v[j] = (_Float16)0.f;
                }
                *(v8h*)&As[astrip][lslot][j0] = v;
            }
        }
        // ---- stage B tile (32 x 64), swizzled ----
        {
            const int gn = blockCol + nB;
            const int kb = kg * 8;
            v8h v;
            if (transB) {
                if (gn < N) {
                    v = *(const v8h*)&B[zB + (long)gn * ldb + k0 + kb];
                } else {
                    #pragma unroll
                    for (int j = 0; j < 8; ++j) v[j] = (_Float16)0.f;
                }
            } else {
                #pragma unroll
                for (int j = 0; j < 8; ++j) {
                    const int gk = k0 + kb + j;
                    v[j] = (gn < N) ? B[zB + (long)gk * ldb + gn] : (_Float16)0.f;
                }
            }
            *(v8h*)&Bs[ctB][laneB][j0B] = v;
        }
        // prefetch next K-tile of A into cache (global_prefetch_b8)
        if (k0 + BK < K && blockRow + ar < M)
            __builtin_prefetch(&A[zA + (long)(blockRow + ar) * lda + k0 + BK], 0, 3);
        __syncthreads();

        // ---- fragments: 2 ds_load_b128 each, then 4 WMMAs ----
        const v16h a0 = frag16(&As[wr + 0][lane][0]);
        const v16h a1 = frag16(&As[wr + 1][lane][0]);
        const v16h b0 = frag16(&Bs[wc + 0][lane][0]);
        const v16h b1 = frag16(&Bs[wc + 1][lane][0]);
        acc[0][0] = wmma_step(a0, b0, acc[0][0]);
        acc[0][1] = wmma_step(a0, b1, acc[0][1]);
        acc[1][0] = wmma_step(a1, b0, acc[1][0]);
        acc[1][1] = wmma_step(a1, b1, acc[1][1]);
        __syncthreads();
    }

    // ---- epilogue: C/D layout: elem r -> row r + 8*(lane>>4), col lane&15 ----
    const int rofs = (lane >> 4) ? 8 : 0;
    const int cl = lane & 15;
    #pragma unroll
    for (int i = 0; i < 2; ++i) {
        #pragma unroll
        for (int j = 0; j < 2; ++j) {
            const int rbase = blockRow + (wr + i) * 16 + rofs;
            const int gc = blockCol + (wc + j) * 16 + cl;
            #pragma unroll
            for (int r = 0; r < 8; ++r) {
                const int gr = rbase + r;
                if (gr < M && gc < N) {
                    float v = acc[i][j][r];
                    long o = zC + (long)gr * ldc + gc;
                    if (flags & GF_BIAS)  v += bias[gc];
                    if (flags & GF_RESID) v += resid[o];
                    if (flags & GF_RELU)  v = fmaxf(v, 0.f);
                    if (C)  C[o]  = v;
                    if (Ch) Ch[o] = (_Float16)v;
                }
            }
        }
    }
}

// ============================ glue kernels =================================
__global__ void k_f32_to_f16(const float* __restrict__ x, _Float16* __restrict__ y, long n) {
    long i = (long)blockIdx.x * blockDim.x + threadIdx.x;
    long st = (long)gridDim.x * blockDim.x;
    for (; i < n; i += st) y[i] = (_Float16)x[i];
}

__global__ void k_add_pe(float* __restrict__ xe, _Float16* __restrict__ xeh, long n) {
    long i = (long)blockIdx.x * blockDim.x + threadIdx.x;
    long st = (long)gridDim.x * blockDim.x;
    for (; i < n; i += st) {
        int h  = (int)(i % HDIM);
        int nn = (int)((i / HDIM) % NPATCH);
        float base = (float)nn / powf(10000.0f, (float)h / (float)HDIM);
        float pe = ((h & 1) == 0) ? sinf(base) : cosf(base);
        float v = xe[i] + pe;
        xe[i] = v; xeh[i] = (_Float16)v;
    }
}

// qkv f32 [NTOK,3H] -> qh/kh/vh f16 [B*NH, NPATCH, DH]
__global__ void k_split_heads(const float* __restrict__ qkv, _Float16* __restrict__ qh,
                              _Float16* __restrict__ kh, _Float16* __restrict__ vh, long n) {
    long i = (long)blockIdx.x * blockDim.x + threadIdx.x;
    long st = (long)gridDim.x * blockDim.x;
    for (; i < n; i += st) {
        int col = (int)(i % (3 * HDIM));
        long tok = i / (3 * HDIM);
        int b = (int)(tok >> 7), nn = (int)(tok & 127);
        int which = col / HDIM, h = (col % HDIM) / DH, d = col % DH;
        long dst = (((long)(b * NHEAD + h)) * NPATCH + nn) * DH + d;
        _Float16 v = (_Float16)qkv[i];
        if (which == 0) qh[dst] = v; else if (which == 1) kh[dst] = v; else vh[dst] = v;
    }
}

// scores [B*NH*NPATCH, NPATCH] f32 -> masked softmax -> probs f16
__global__ void k_softmax(const float* __restrict__ s, const int* __restrict__ mask,
                          _Float16* __restrict__ p) {
    __shared__ float buf[NPATCH];
    int row = blockIdx.x, c = threadIdx.x;
    int b = row >> 10;  // row / (NH*NPATCH)
    float v = s[(long)row * NPATCH + c] * 0.125f;  // 1/sqrt(DH)
    if (mask[b * NPATCH + c] == 0) v = -1e30f;
    buf[c] = v; __syncthreads();
    for (int stp = 64; stp > 0; stp >>= 1) { if (c < stp) buf[c] = fmaxf(buf[c], buf[c + stp]); __syncthreads(); }
    float mx = buf[0]; __syncthreads();
    float e = __expf(v - mx);
    buf[c] = e; __syncthreads();
    for (int stp = 64; stp > 0; stp >>= 1) { if (c < stp) buf[c] += buf[c + stp]; __syncthreads(); }
    p[(long)row * NPATCH + c] = (_Float16)(e / buf[0]);
}

// ao f32 [B*NH, NPATCH, DH] -> attn f16 [NTOK, H]
__global__ void k_merge_heads(const float* __restrict__ ao, _Float16* __restrict__ attn, long n) {
    long i = (long)blockIdx.x * blockDim.x + threadIdx.x;
    long st = (long)gridDim.x * blockDim.x;
    for (; i < n; i += st) {
        int col = (int)(i % HDIM);
        long tok = i / HDIM;
        int b = (int)(tok >> 7), nn = (int)(tok & 127);
        int h = col / DH, d = col % DH;
        attn[i] = (_Float16)ao[(((long)(b * NHEAD + h)) * NPATCH + nn) * DH + d];
    }
}

// y = LN(x) * g + b per row of C columns; optional f16 copy
__global__ void k_layernorm(const float* __restrict__ x, const float* __restrict__ g,
                            const float* __restrict__ bta, float* __restrict__ y,
                            _Float16* __restrict__ yh, int C) {
    __shared__ float rs[256], rq[256];
    int row = blockIdx.x, tid = threadIdx.x;
    const float* xr = x + (long)row * C;
    float s = 0.f, ss = 0.f;
    for (int c = tid; c < C; c += 256) { float v = xr[c]; s += v; ss += v * v; }
    rs[tid] = s; rq[tid] = ss; __syncthreads();
    for (int stp = 128; stp > 0; stp >>= 1) {
        if (tid < stp) { rs[tid] += rs[tid + stp]; rq[tid] += rq[tid + stp]; }
        __syncthreads();
    }
    float mean = rs[0] / C;
    float var  = rq[0] / C - mean * mean;
    float inv  = rsqrtf(var + 1e-5f);
    for (int c = tid; c < C; c += 256) {
        float v = (xr[c] - mean) * inv * g[c] + bta[c];
        y[(long)row * C + c] = v;
        if (yh) yh[(long)row * C + c] = (_Float16)v;
    }
}

// causal conv: in [B,T,Cin], w [Cout,Cin,Kk] (OIH), out [B,T,Cout]
__global__ void k_causal_conv(const float* __restrict__ in, const float* __restrict__ w,
                              const float* __restrict__ bias, float* __restrict__ out,
                              int T, int Cin, int Cout, int Kk, int pad, int relu, long n) {
    long i = (long)blockIdx.x * blockDim.x + threadIdx.x;
    long st = (long)gridDim.x * blockDim.x;
    for (; i < n; i += st) {
        int co = (int)(i % Cout);
        int t  = (int)((i / Cout) % T);
        int b  = (int)(i / ((long)Cout * T));
        float s = bias[co];
        for (int kk = 0; kk < Kk; ++kk) {
            int tt = t + kk - pad;
            if (tt < 0 || tt >= T) continue;
            const float* ip = in + ((long)b * T + tt) * Cin;
            const float* wp = w + ((long)co * Cin) * Kk + kk;
            for (int ci = 0; ci < Cin; ++ci) s += wp[(long)ci * Kk] * ip[ci];
        }
        if (relu) s = fmaxf(s, 0.f);
        out[i] = s;
    }
}

__global__ void k_relu_add(const float* __restrict__ a, const float* __restrict__ b,
                           float* __restrict__ y, _Float16* __restrict__ yh, long n) {
    long i = (long)blockIdx.x * blockDim.x + threadIdx.x;
    long st = (long)gridDim.x * blockDim.x;
    for (; i < n; i += st) {
        float v = fmaxf(a[i] + b[i], 0.f);
        y[i] = v; yh[i] = (_Float16)v;
    }
}

__global__ void k_masked_mean(const float* __restrict__ tcn, const int* __restrict__ mask,
                              float* __restrict__ gf, _Float16* __restrict__ gfh, long n) {
    long i = (long)blockIdx.x * blockDim.x + threadIdx.x;
    long st = (long)gridDim.x * blockDim.x;
    for (; i < n; i += st) {
        int c = (int)(i % THDIM), b = (int)(i / THDIM);
        float s = 0.f;
        for (int nn = 0; nn < NPATCH; ++nn)
            if (mask[b * NPATCH + nn]) s += tcn[((long)(b * NPATCH + nn)) * THDIM + c];
        float v = s / (float)NPATCH;  // torch.mean divides by N
        gf[i] = v; gfh[i] = (_Float16)v;
    }
}

__global__ void k_tanh_f16(const float* __restrict__ x, _Float16* __restrict__ y, long n) {
    long i = (long)blockIdx.x * blockDim.x + threadIdx.x;
    long st = (long)gridDim.x * blockDim.x;
    for (; i < n; i += st) y[i] = (_Float16)tanhf(x[i]);
}

// A1 = relu(tanh(t1-t2))*mask_r*mask_c + I
__global__ void k_adj_build(const float* __restrict__ t1, const float* __restrict__ t2,
                            const int* __restrict__ mask, float* __restrict__ A1, long n) {
    long i = (long)blockIdx.x * blockDim.x + threadIdx.x;
    long st = (long)gridDim.x * blockDim.x;
    for (; i < n; i += st) {
        int b = (int)(i >> 14), r = (int)((i >> 7) & 127), c = (int)(i & 127);
        float a = tanhf(t1[i] - t2[i]);
        float e = fmaxf(a, 0.f);
        if (mask[b * NPATCH + r] == 0 || mask[b * NPATCH + c] == 0) e = 0.f;
        if (r == c) e += 1.0f;
        A1[i] = e;
    }
}

// dinv[b,j] = rsqrt(sum_i A1[b,i,j])
__global__ void k_colsum_rsqrt(const float* __restrict__ A1, float* __restrict__ dinv, long n) {
    long i = (long)blockIdx.x * blockDim.x + threadIdx.x;
    long st = (long)gridDim.x * blockDim.x;
    for (; i < n; i += st) {
        int b = (int)(i >> 7), j = (int)(i & 127);
        float s = 0.f;
        for (int r = 0; r < NPATCH; ++r) s += A1[((long)(b * NPATCH + r)) * NPATCH + j];
        dinv[i] = rsqrtf(s);
    }
}

// AnT[b,j,i] = dinv[b,i]*dinv[b,j]*A1[b,i,j]  (pre-transposed for source->target agg)
__global__ void k_norm_T(const float* __restrict__ A1, const float* __restrict__ dinv,
                         _Float16* __restrict__ AnT, long n) {
    long i = (long)blockIdx.x * blockDim.x + threadIdx.x;
    long st = (long)gridDim.x * blockDim.x;
    for (; i < n; i += st) {
        int b = (int)(i >> 14), j = (int)((i >> 7) & 127), k = (int)(i & 127);
        AnT[i] = (_Float16)(dinv[b * NPATCH + k] * dinv[b * NPATCH + j] *
                            A1[((long)(b * NPATCH + k)) * NPATCH + j]);
    }
}

// ef_h = f16(ef * mask_row)
__global__ void k_mask_rows_f16(const float* __restrict__ x, const int* __restrict__ mask,
                                _Float16* __restrict__ y, int C, long n) {
    long i = (long)blockIdx.x * blockDim.x + threadIdx.x;
    long st = (long)gridDim.x * blockDim.x;
    for (; i < n; i += st) {
        long tok = i / C;
        float v = mask[tok] ? x[i] : 0.f;
        y[i] = (_Float16)v;
    }
}

__global__ void k_add2(const float* __restrict__ a, const float* __restrict__ b,
                       float* __restrict__ y, _Float16* __restrict__ yh, long n) {
    long i = (long)blockIdx.x * blockDim.x + threadIdx.x;
    long st = (long)gridDim.x * blockDim.x;
    for (; i < n; i += st) { float v = a[i] + b[i]; y[i] = v; yh[i] = (_Float16)v; }
}

// ============================ host side ====================================
struct Arena { char* p; size_t off; };
static void* araw(Arena& a, size_t bytes) {
    size_t al = (bytes + 255) & ~(size_t)255;
    void* r = a.p + a.off; a.off += al; return r;
}
#define AF32(a, n) ((float*)araw((a), (size_t)(n) * 4))
#define AF16(a, n) ((_Float16*)araw((a), (size_t)(n) * 2))

static void gemm(hipStream_t s, const _Float16* A, const _Float16* B, float* C,
                 _Float16* Ch, const float* bias, const float* resid,
                 int M, int N, int K, int lda, int ldb, int ldc,
                 long sA, long sB, long sC, int batch, int transB, int flags) {
    dim3 g((N + 63) / 64, (M + 127) / 128, batch);
    wmma_gemm<<<g, dim3(256), 0, s>>>(A, B, C, Ch, bias, resid,
                                      M, N, K, lda, ldb, ldc, sA, sB, sC, transB, flags);
}
#define EW(kern, n, ...) kern<<<dim3((unsigned)(((n) + 255) / 256)), dim3(256), 0, stream>>>(__VA_ARGS__)

struct XfP { const float *wqkv,*bqkv,*wo,*bo,*ln1g,*ln1b,*ln2g,*ln2b,*w1,*b1,*w2,*b2; };

extern "C" void kernel_launch(void* const* d_in, const int* in_sizes, int n_in,
                              void* d_out, int out_size, void* d_ws, size_t ws_size,
                              hipStream_t stream) {
    (void)in_sizes; (void)n_in; (void)out_size; (void)ws_size;
    int ix = 0;
    auto F = [&]() { return (const float*)d_in[ix++]; };
    const float* x    = F();
    const int*   mask = (const int*)d_in[ix++];
    const float* patch_w = F(); const float* patch_b = F();
    auto RX = [&]() { XfP p; p.wqkv=F(); p.bqkv=F(); p.wo=F(); p.bo=F();
                      p.ln1g=F(); p.ln1b=F(); p.ln2g=F(); p.ln2b=F();
                      p.w1=F(); p.b1=F(); p.w2=F(); p.b2=F(); return p; };
    XfP enc = RX(), dec = RX();
    const float *tw1=F(), *tb1=F(), *tw2=F(), *tb2=F(), *twd=F(), *tbd=F();
    const float *fw1=F(), *fb1=F(), *fw2=F(), *fb2=F();
    const float *m1w1=F(), *m1b1=F(), *m1w2=F(), *m1b2=F();
    const float *m2w1=F(), *m2b1=F(), *m2w2=F(), *m2b2=F();
    const float *gw1=F(), *gb1=F(), *gw2=F(), *gb2=F();
    const float *out_w=F(), *out_b=F();

    Arena ws{(char*)d_ws, 0};
    auto toh = [&](const float* w, size_t n) {
        _Float16* p = AF16(ws, n);
        EW(k_f32_to_f16, n, w, p, (long)n);
        return p;
    };

    // ---- weights -> f16 ----
    _Float16* patchW = toh(patch_w, 128 * HDIM);
    _Float16 *eQKV = toh(enc.wqkv, (size_t)NLAYER*HDIM*3*HDIM), *eWO = toh(enc.wo, (size_t)NLAYER*HDIM*HDIM);
    _Float16 *eW1 = toh(enc.w1, (size_t)NLAYER*HDIM*FFDIM),     *eW2 = toh(enc.w2, (size_t)NLAYER*FFDIM*HDIM);
    _Float16 *dQKV = toh(dec.wqkv, (size_t)NLAYER*HDIM*3*HDIM), *dWO = toh(dec.wo, (size_t)NLAYER*HDIM*HDIM);
    _Float16 *dW1 = toh(dec.w1, (size_t)NLAYER*HDIM*FFDIM),     *dW2 = toh(dec.w2, (size_t)NLAYER*FFDIM*HDIM);
    _Float16 *FW1 = toh(fw1, THDIM*2*THDIM),  *FW2 = toh(fw2, (size_t)2*THDIM*HDIM);
    _Float16 *MA1 = toh(m1w1, THDIM*2*THDIM), *MA2 = toh(m1w2, (size_t)2*THDIM*NN2);
    _Float16 *MB1 = toh(m2w1, THDIM*2*THDIM), *MB2 = toh(m2w2, (size_t)2*THDIM*NN2);
    _Float16 *GW1 = toh(gw1, HDIM*THDIM),     *GW2 = toh(gw2, THDIM*HDIM);
    _Float16 *OW  = toh(out_w, HDIM*FEATD);

    // ---- activations ----
    _Float16* xh   = AF16(ws, (size_t)NTOK * 128);             // patched input
    float*    xe   = AF32(ws, (size_t)NTOK * HDIM);            // encoder stream
    _Float16* xeh  = AF16(ws, (size_t)NTOK * HDIM);
    float*    qkv  = AF32(ws, (size_t)NTOK * 3 * HDIM);
    _Float16 *qh = AF16(ws, (size_t)BB*NHEAD*NPATCH*DH), *kh = AF16(ws, (size_t)BB*NHEAD*NPATCH*DH),
             *vh = AF16(ws, (size_t)BB*NHEAD*NPATCH*DH);
    float*    scores = AF32(ws, (size_t)BB*NHEAD*NPATCH*NPATCH);
    _Float16* probs  = AF16(ws, (size_t)BB*NHEAD*NPATCH*NPATCH);
    float*    ao     = AF32(ws, (size_t)BB*NHEAD*NPATCH*DH);
    _Float16* attn   = AF16(ws, (size_t)NTOK*HDIM);
    float*    ybuf   = AF32(ws, (size_t)NTOK*HDIM);
    float*    ln1    = AF32(ws, (size_t)NTOK*HDIM);
    _Float16* ln1h   = AF16(ws, (size_t)NTOK*HDIM);
    _Float16* ffh    = AF16(ws, (size_t)NTOK*FFDIM);
    float*    y2     = AF32(ws, (size_t)NTOK*HDIM);

    float *h1 = AF32(ws, (size_t)NTOK*THDIM), *h2 = AF32(ws, (size_t)NTOK*THDIM),
          *rsd = AF32(ws, (size_t)NTOK*THDIM), *tcn = AF32(ws, (size_t)NTOK*THDIM);
    _Float16* tcnh = AF16(ws, (size_t)NTOK*THDIM);
    float* gf = AF32(ws, (size_t)BB*THDIM); _Float16* gfh = AF16(ws, (size_t)BB*THDIM);
    _Float16* mhid = AF16(ws, (size_t)BB*2*THDIM);
    float* mraw = AF32(ws, (size_t)BB*NN2);
    _Float16 *m1h = AF16(ws, (size_t)BB*NN2), *m2h = AF16(ws, (size_t)BB*NN2);
    float *t1 = AF32(ws, (size_t)BB*NN2), *t2 = AF32(ws, (size_t)BB*NN2), *A1 = AF32(ws, (size_t)BB*NN2);
    float* dinv = AF32(ws, (size_t)BB*NPATCH);
    _Float16* AnT = AF16(ws, (size_t)BB*NN2);
    float* ef = AF32(ws, (size_t)NTOK*HDIM); _Float16* efh = AF16(ws, (size_t)NTOK*HDIM);
    _Float16* X1 = AF16(ws, (size_t)NTOK*THDIM);
    float* g1 = AF32(ws, (size_t)NTOK*THDIM); _Float16* g1h = AF16(ws, (size_t)NTOK*THDIM);
    _Float16* X2 = AF16(ws, (size_t)NTOK*HDIM);
    float* g2 = AF32(ws, (size_t)NTOK*HDIM);
    float* comb = AF32(ws, (size_t)NTOK*HDIM); _Float16* combh = AF16(ws, (size_t)NTOK*HDIM);

    const long HB = (long)NPATCH * DH;        // per-head A/B stride
    const long SB = (long)NPATCH * NPATCH;    // per-head score stride

    auto xformer = [&](const XfP& p, const _Float16* Wqkv, const _Float16* Wo,
                       const _Float16* W1, const _Float16* W2, float* sf32, _Float16* sf16) {
        for (int l = 0; l < NLAYER; ++l) {
            gemm(stream, sf16, Wqkv + (long)l*HDIM*3*HDIM, qkv, nullptr,
                 p.bqkv + (long)l*3*HDIM, nullptr,
                 NTOK, 3*HDIM, HDIM, HDIM, 3*HDIM, 3*HDIM, 0, 0, 0, 1, 0, GF_BIAS);
            EW(k_split_heads, (long)NTOK*3*HDIM, qkv, qh, kh, vh, (long)NTOK*3*HDIM);
            gemm(stream, qh, kh, scores, nullptr, nullptr, nullptr,
                 NPATCH, NPATCH, DH, DH, DH, NPATCH, HB, HB, SB, BB*NHEAD, 1, 0);
            k_softmax<<<dim3(BB*NHEAD*NPATCH), dim3(NPATCH), 0, stream>>>(scores, mask, probs);
            gemm(stream, probs, vh, ao, nullptr, nullptr, nullptr,
                 NPATCH, DH, NPATCH, NPATCH, DH, DH, SB, HB, HB, BB*NHEAD, 0, 0);
            EW(k_merge_heads, (long)NTOK*HDIM, ao, attn, (long)NTOK*HDIM);
            gemm(stream, attn, Wo + (long)l*HDIM*HDIM, ybuf, nullptr,
                 p.bo + (long)l*HDIM, sf32,
                 NTOK, HDIM, HDIM, HDIM, HDIM, HDIM, 0, 0, 0, 1, 0, GF_BIAS | GF_RESID);
            k_layernorm<<<dim3(NTOK), dim3(256), 0, stream>>>(
                ybuf, p.ln1g + (long)l*HDIM, p.ln1b + (long)l*HDIM, ln1, ln1h, HDIM);
            gemm(stream, ln1h, W1 + (long)l*HDIM*FFDIM, nullptr, ffh,
                 p.b1 + (long)l*FFDIM, nullptr,
                 NTOK, FFDIM, HDIM, HDIM, FFDIM, FFDIM, 0, 0, 0, 1, 0, GF_BIAS | GF_RELU);
            gemm(stream, ffh, W2 + (long)l*FFDIM*HDIM, y2, nullptr,
                 p.b2 + (long)l*HDIM, ln1,
                 NTOK, HDIM, FFDIM, FFDIM, HDIM, HDIM, 0, 0, 0, 1, 0, GF_BIAS | GF_RESID);
            k_layernorm<<<dim3(NTOK), dim3(256), 0, stream>>>(
                y2, p.ln2g + (long)l*HDIM, p.ln2b + (long)l*HDIM, sf32, sf16, HDIM);
        }
    };

    // ---- patch embedding + PE ----
    EW(k_f32_to_f16, (long)NTOK*128, x, xh, (long)NTOK*128);   // [B,SEQ,FEAT]==[NTOK,128]
    gemm(stream, xh, patchW, xe, nullptr, patch_b, nullptr,
         NTOK, HDIM, 128, 128, HDIM, HDIM, 0, 0, 0, 1, 0, GF_BIAS);
    EW(k_add_pe, (long)NTOK*HDIM, xe, xeh, (long)NTOK*HDIM);

    // ---- encoder ----
    xformer(enc, eQKV, eWO, eW1, eW2, xe, xeh);                // xe/xeh now hold enc

    // ---- TCN (DynamicGraphLearner) ----
    EW(k_causal_conv, (long)NTOK*THDIM, xe, tw1, tb1, h1, NPATCH, HDIM,  THDIM, 3, 2, 1, (long)NTOK*THDIM);
    EW(k_causal_conv, (long)NTOK*THDIM, h1, tw2, tb2, h2, NPATCH, THDIM, THDIM, 3, 2, 1, (long)NTOK*THDIM);
    EW(k_causal_conv, (long)NTOK*THDIM, xe, twd, tbd, rsd, NPATCH, HDIM, THDIM, 1, 0, 0, (long)NTOK*THDIM);
    EW(k_relu_add, (long)NTOK*THDIM, h2, rsd, tcn, tcnh, (long)NTOK*THDIM);
    EW(k_masked_mean, (long)BB*THDIM, tcn, mask, gf, gfh, (long)BB*THDIM);

    // ---- dynamic adjacency ----
    gemm(stream, gfh, MA1, nullptr, mhid, m1b1, nullptr, BB, 2*THDIM, THDIM,
         THDIM, 2*THDIM, 2*THDIM, 0, 0, 0, 1, 0, GF_BIAS | GF_RELU);
    gemm(stream, mhid, MA2, mraw, nullptr, m1b2, nullptr, BB, NN2, 2*THDIM,
         2*THDIM, NN2, NN2, 0, 0, 0, 1, 0, GF_BIAS);
    EW(k_tanh_f16, (long)BB*NN2, mraw, m1h, (long)BB*NN2);
    gemm(stream, gfh, MB1, nullptr, mhid, m2b1, nullptr, BB, 2*THDIM, THDIM,
         THDIM, 2*THDIM, 2*THDIM, 0, 0, 0, 1, 0, GF_BIAS | GF_RELU);
    gemm(stream, mhid, MB2, mraw, nullptr, m2b2, nullptr, BB, NN2, 2*THDIM,
         2*THDIM, NN2, NN2, 0, 0, 0, 1, 0, GF_BIAS);
    EW(k_tanh_f16, (long)BB*NN2, mraw, m2h, (long)BB*NN2);
    gemm(stream, m1h, m2h, t1, nullptr, nullptr, nullptr, NPATCH, NPATCH, NPATCH,
         NPATCH, NPATCH, NPATCH, NN2, NN2, NN2, BB, 1, 0);
    gemm(stream, m2h, m1h, t2, nullptr, nullptr, nullptr, NPATCH, NPATCH, NPATCH,
         NPATCH, NPATCH, NPATCH, NN2, NN2, NN2, BB, 1, 0);
    EW(k_adj_build, (long)BB*NN2, t1, t2, mask, A1, (long)BB*NN2);
    EW(k_colsum_rsqrt, (long)BB*NPATCH, A1, dinv, (long)BB*NPATCH);
    EW(k_norm_T, (long)BB*NN2, A1, dinv, AnT, (long)BB*NN2);

    // ---- node features (mlpf) ----
    gemm(stream, tcnh, FW1, nullptr, ffh, fb1, nullptr, NTOK, 2*THDIM, THDIM,
         THDIM, 2*THDIM, 2*THDIM, 0, 0, 0, 1, 0, GF_BIAS | GF_RELU);
    gemm(stream, ffh, FW2, ef, nullptr, fb2, nullptr, NTOK, HDIM, 2*THDIM,
         2*THDIM, HDIM, HDIM, 0, 0, 0, 1, 0, GF_BIAS);
    EW(k_mask_rows_f16, (long)NTOK*HDIM, ef, mask, efh, HDIM, (long)NTOK*HDIM);

    // ---- GCN (2 layers, AnT aggregation) ----
    gemm(stream, efh, GW1, nullptr, X1, gb1, nullptr, NTOK, THDIM, HDIM,
         HDIM, THDIM, THDIM, 0, 0, 0, 1, 0, GF_BIAS);
    gemm(stream, AnT, X1, g1, g1h, nullptr, nullptr, NPATCH, THDIM, NPATCH,
         NPATCH, THDIM, THDIM, NN2, (long)NPATCH*THDIM, (long)NPATCH*THDIM, BB, 0, GF_RELU);
    gemm(stream, g1h, GW2, nullptr, X2, gb2, nullptr, NTOK, HDIM, THDIM,
         THDIM, HDIM, HDIM, 0, 0, 0, 1, 0, GF_BIAS);
    gemm(stream, AnT, X2, g2, nullptr, nullptr, nullptr, NPATCH, HDIM, NPATCH,
         NPATCH, HDIM, HDIM, NN2, (long)NPATCH*HDIM, (long)NPATCH*HDIM, BB, 0, GF_RELU);

    // ---- combine + decoder + output ----
    EW(k_add2, (long)NTOK*HDIM, xe, g2, comb, combh, (long)NTOK*HDIM);
    xformer(dec, dQKV, dWO, dW1, dW2, comb, combh);
    gemm(stream, combh, OW, (float*)d_out, nullptr, out_b, nullptr,
         NTOK, FEATD, HDIM, HDIM, FEATD, FEATD, 0, 0, 0, 1, 0, GF_BIAS);
}
